// BiasedAxialAttention_18330920419589
// MI455X (gfx1250) — matrix operand order repaired and checked by reading, not compiled, not run on hardware
//
#include <hip/hip_runtime.h>

// ---------------------------------------------------------------------------
// BiasedAxialAttention for MI455X (gfx1250, wave32, WMMA f16->f32)
// Shapes: B=1, L=384, D_PAIR=D_BIAS=128, H=4, Dh=32 (H*Dh=128)
// ---------------------------------------------------------------------------

#define Lq   384
#define DP   128
#define NH   4
#define DH   32
#define KBIG (Lq * DH)          // 12288 : contraction length of logits einsum
#define NROW ((long)Lq * Lq)    // 147456 rows of the (n,i) flattened pair

typedef __attribute__((ext_vector_type(16))) _Float16 v16h;
typedef __attribute__((ext_vector_type(8)))  float    v8f;

union Frag16 {                   // 16 halves = 8 VGPRs, also viewable as 2x16B
  v16h v;
  _Float16 h[16];
  uint4 q[2];
};

// A-matrix 16x32 f16 fragment from row-major storage (contiguous K).
// lanes 0-15: M=lane, K = {0..7, 16..23}; lanes 16-31: M=lane-16, K={8..15,24..31}
__device__ __forceinline__ v16h loadA(const _Float16* __restrict__ base, long lda) {
  int lane = threadIdx.x & 31;
  const _Float16* p = base + (long)(lane & 15) * lda + ((lane >> 4) << 3);
  Frag16 a;
  a.q[0] = *(const uint4*)(p);        // K = kb .. kb+7
  a.q[1] = *(const uint4*)(p + 16);   // K = kb+16 .. kb+23
  return a.v;
}

__device__ __forceinline__ v8f wmma_f16(v16h a, v16h b, v8f c) {
  return __builtin_amdgcn_wmma_f32_16x16x32_f16(false, a, false, b, (short)0, c,
                                                false, false);
}

__device__ __forceinline__ float wave_sum(float s) {
  #pragma unroll
  for (int m = 16; m >= 1; m >>= 1) s += __shfl_xor(s, m, 32);
  return s;
}
__device__ __forceinline__ float wave_max(float s) {
  #pragma unroll
  for (int m = 16; m >= 1; m >>= 1) s = fmaxf(s, __shfl_xor(s, m, 32));
  return s;
}

// ---------------------------------------------------------------------------
// 1) LayerNorm of transposed pair -> f16 P.  row = n*L + i reads pair[i][n][:]
// ---------------------------------------------------------------------------
__global__ void k_ln_pair(const float* __restrict__ pair,
                          const float* __restrict__ g, const float* __restrict__ bta,
                          _Float16* __restrict__ P) {
  int lane = threadIdx.x & 31;
  long row = (long)blockIdx.x * 8 + (threadIdx.x >> 5);
  int n = (int)(row / Lq), i = (int)(row % Lq);
  const float* src = pair + ((long)i * Lq + n) * DP;
  float x[4];
  #pragma unroll
  for (int t = 0; t < 4; ++t) x[t] = src[t * 32 + lane];
  float mean = wave_sum(x[0] + x[1] + x[2] + x[3]) * (1.0f / DP);
  float vv = 0.f;
  #pragma unroll
  for (int t = 0; t < 4; ++t) { float d = x[t] - mean; vv += d * d; }
  float rstd = rsqrtf(wave_sum(vv) * (1.0f / DP) + 1e-5f);
  _Float16* dst = P + row * DP;
  #pragma unroll
  for (int t = 0; t < 4; ++t) {
    int c = t * 32 + lane;
    dst[c] = (_Float16)((x[t] - mean) * rstd * g[c] + bta[c]);
  }
}

// ---------------------------------------------------------------------------
// 2) LN(bias transposed) @ Wb  -> bprj[(i*L + j)*4 + h]
// ---------------------------------------------------------------------------
__global__ void k_ln_bias(const float* __restrict__ bias,
                          const float* __restrict__ g, const float* __restrict__ bta,
                          const float* __restrict__ Wb,
                          float* __restrict__ bprj) {
  int lane = threadIdx.x & 31;
  long row = (long)blockIdx.x * 8 + (threadIdx.x >> 5);  // row = i*L + j
  int i = (int)(row / Lq), j = (int)(row % Lq);
  const float* src = bias + ((long)j * Lq + i) * DP;
  float x[4];
  #pragma unroll
  for (int t = 0; t < 4; ++t) x[t] = src[t * 32 + lane];
  float mean = wave_sum(x[0] + x[1] + x[2] + x[3]) * (1.0f / DP);
  float vv = 0.f;
  #pragma unroll
  for (int t = 0; t < 4; ++t) { float d = x[t] - mean; vv += d * d; }
  float rstd = rsqrtf(wave_sum(vv) * (1.0f / DP) + 1e-5f);
  float acc[NH] = {0.f, 0.f, 0.f, 0.f};
  #pragma unroll
  for (int t = 0; t < 4; ++t) {
    int c = t * 32 + lane;
    float ln = (x[t] - mean) * rstd * g[c] + bta[c];
    #pragma unroll
    for (int h = 0; h < NH; ++h) acc[h] += ln * Wb[c * NH + h];
  }
  #pragma unroll
  for (int h = 0; h < NH; ++h) {
    float s = wave_sum(acc[h]);
    if (lane == 0) bprj[row * NH + h] = s;
  }
}

// ---------------------------------------------------------------------------
// 3) Projection GEMM:  [147456,128] x [128,128] via WMMA.
//    W staged TRANSPOSED in LDS (Wl[n*128+k], f16): B-fragments are two
//    contiguous ds_load_b128 per k-step, shared by two 16-row A tiles.
//    Tiles are 32 rows and 32 | 384 so n = row0/384 is block-uniform (no
//    per-lane division in the epilogue).
//    mode 0: Q (x 1/sqrt(32))  -> Qs[h][i][n*32+d]     (k-contiguous)
//    mode 1: K (x 1/L)         -> Ks[h][j][n*32+d]     (k-contiguous)
//    mode 2: V                 -> Vt[h][d][n][j]       (j-contiguous)
//    mode 3: gate = sigmoid(x+bg), f32 natural layout [n*L+i][c]
// ---------------------------------------------------------------------------
__global__ void k_proj(const _Float16* __restrict__ P,
                       const float* __restrict__ W,
                       const float* __restrict__ bias128,
                       void* __restrict__ dst, int mode) {
  __shared__ _Float16 Wl[DP * DP];           // transposed: Wl[n*128 + k]
  for (int t = threadIdx.x; t < DP * DP; t += blockDim.x) {
    int k = t >> 7, n = t & 127;             // W[t] = W[k][n], coalesced read
    Wl[n * DP + k] = (_Float16)W[t];
  }
  __syncthreads();

  int lane = threadIdx.x & 31;
  long row0 = (long)blockIdx.x * 32;          // 4608 blocks over M
  int n0 = (threadIdx.x >> 5) * 16;           // 8 waves cover N = 128
  const _Float16* wb = Wl + (long)(n0 + (lane & 15)) * DP + ((lane >> 4) << 3);

  v8f acc0 = {}, acc1 = {};
  #pragma unroll
  for (int kk = 0; kk < DP; kk += 32) {
    v16h a0 = loadA(P + row0 * DP + kk, DP);
    v16h a1 = loadA(P + (row0 + 16) * DP + kk, DP);
    Frag16 b;
    b.q[0] = *(const uint4*)(wb + kk);        // ds_load_b128
    b.q[1] = *(const uint4*)(wb + kk + 16);
    acc0 = wmma_f16(a0, b.v, acc0);
    acc1 = wmma_f16(a1, b.v, acc1);
  }

  // block-uniform n index, per-lane small adds only
  const int nidx  = (int)(row0 / Lq);
  const int ibase = (int)(row0 % Lq);
  const int col = n0 + (lane & 15);           // col = h*32 + d
  const int h = col >> 5, d = col & 31;

  #pragma unroll
  for (int half = 0; half < 2; ++half) {
    v8f acc = half ? acc1 : acc0;
    #pragma unroll
    for (int r = 0; r < 8; ++r) {
      int iidx = ibase + half * 16 + r + ((lane >> 4) << 3);
      float val = acc[r];
      if (mode == 0) {
        ((_Float16*)dst)[((long)(h * Lq + iidx)) * KBIG + nidx * DH + d] =
            (_Float16)(val * 0.17677669529663687f);     // 1/sqrt(32)
      } else if (mode == 1) {
        ((_Float16*)dst)[((long)(h * Lq + iidx)) * KBIG + nidx * DH + d] =
            (_Float16)(val * (1.0f / (float)Lq));
      } else if (mode == 2) {
        // Vt[h][d][n][j] : j-contiguous for the attn@V B-fragments
        ((_Float16*)dst)[(((long)(h * DH + d) * Lq + nidx) * Lq) + iidx] =
            (_Float16)val;
      } else {
        float z = val + bias128[col];
        ((float*)dst)[((long)nidx * Lq + iidx) * DP + col] =
            1.0f / (1.0f + __expf(-z));
      }
    }
  }
}

// ---------------------------------------------------------------------------
// 4) logits[h][i][j] = sum_k Qs[h][i][k] * Ks[h][j][k]  (K = 12288) + bprj
//    one wave per 32x16 (i,j) tile: two accumulators share each B fragment.
// ---------------------------------------------------------------------------
__global__ void k_logits(const _Float16* __restrict__ Qs,
                         const _Float16* __restrict__ Ks,
                         const float* __restrict__ bprj,
                         float* __restrict__ logits) {
  int gw = blockIdx.x * 8 + (threadIdx.x >> 5);   // 0..1151
  int lane = threadIdx.x & 31;
  int jt = gw % 24;
  int r1 = gw / 24;
  int it2 = r1 % 12;          // 32-row i tile
  int h   = r1 / 12;

  const _Float16* Qh = Qs + (long)h * Lq * KBIG;
  const _Float16* Kh = Ks + (long)h * Lq * KBIG;
  const _Float16* pa0 = Qh + (long)(it2 * 32      + (lane & 15)) * KBIG + ((lane >> 4) << 3);
  const _Float16* pa1 = Qh + (long)(it2 * 32 + 16 + (lane & 15)) * KBIG + ((lane >> 4) << 3);
  const _Float16* pb  = Kh + (long)(jt * 16       + (lane & 15)) * KBIG + ((lane >> 4) << 3);

  v8f acc0 = {}, acc1 = {};
  for (int kk = 0; kk < KBIG; kk += 32) {
    __builtin_prefetch(pa0 + kk + 512, 0, 0);
    __builtin_prefetch(pa1 + kk + 512, 0, 0);
    __builtin_prefetch(pb + kk + 512, 0, 0);
    Frag16 a0, a1, b;
    a0.q[0] = *(const uint4*)(pa0 + kk);
    a0.q[1] = *(const uint4*)(pa0 + kk + 16);
    a1.q[0] = *(const uint4*)(pa1 + kk);
    a1.q[1] = *(const uint4*)(pa1 + kk + 16);
    b.q[0]  = *(const uint4*)(pb + kk);
    b.q[1]  = *(const uint4*)(pb + kk + 16);
    acc0 = wmma_f16(a0.v, b.v, acc0);
    acc1 = wmma_f16(a1.v, b.v, acc1);
  }

  int j = jt * 16 + (lane & 15);
  #pragma unroll
  for (int r = 0; r < 8; ++r) {
    int i0 = it2 * 32      + r + ((lane >> 4) << 3);
    int i1 = it2 * 32 + 16 + r + ((lane >> 4) << 3);
    logits[(long)h * Lq * Lq + (long)i0 * Lq + j] =
        acc0[r] + bprj[((long)i0 * Lq + j) * NH + h];
    logits[(long)h * Lq * Lq + (long)i1 * Lq + j] =
        acc1[r] + bprj[((long)i1 * Lq + j) * NH + h];
  }
}

// ---------------------------------------------------------------------------
// 5) softmax over j per (h,i) row; result f16 row-major (A matrix of attn@V)
// ---------------------------------------------------------------------------
__global__ void k_softmax(const float* __restrict__ logits,
                          _Float16* __restrict__ attn) {
  int lane = threadIdx.x & 31;
  long gw = (long)blockIdx.x * 8 + (threadIdx.x >> 5);  // 0..NH*L-1
  const float* src = logits + gw * Lq;
  float x[12];
  float mx = -3.0e38f;
  #pragma unroll
  for (int t = 0; t < 12; ++t) { x[t] = src[t * 32 + lane]; mx = fmaxf(mx, x[t]); }
  mx = wave_max(mx);
  float s = 0.f;
  #pragma unroll
  for (int t = 0; t < 12; ++t) { x[t] = __expf(x[t] - mx); s += x[t]; }
  float inv = 1.0f / wave_sum(s);
  _Float16* dst = attn + gw * Lq;
  #pragma unroll
  for (int t = 0; t < 12; ++t) dst[t * 32 + lane] = (_Float16)(x[t] * inv);
}

// ---------------------------------------------------------------------------
// 6) o[n,i,h,d] = sum_j attn[h][i][j] * Vt[h][d][n][j]; fused gate multiply.
//    wave tile: M=i(16) x N=d(32) (both d sub-tiles share the A fragment).
//    All fragments are contiguous 16B loads thanks to Vt layout.
// ---------------------------------------------------------------------------
__global__ void k_av(const _Float16* __restrict__ attn,
                     const _Float16* __restrict__ Vt,
                     const float* __restrict__ gate,
                     _Float16* __restrict__ go) {
  long gw = (long)blockIdx.x * 8 + (threadIdx.x >> 5);  // 0..36863
  int lane = threadIdx.x & 31;
  int it = (int)(gw % 24); long r1 = gw / 24;
  int n = (int)(r1 % Lq); int h = (int)(r1 / Lq);

  const _Float16* pa = attn + (long)h * Lq * Lq
                     + (long)(it * 16 + (lane & 15)) * Lq + ((lane >> 4) << 3);
  int dl = lane & 15;
  int kb16 = (lane >> 4) << 3;
  const _Float16* pb0 = Vt + ((long)(h * DH + dl)      * Lq + n) * Lq + kb16;
  const _Float16* pb1 = Vt + ((long)(h * DH + 16 + dl) * Lq + n) * Lq + kb16;

  v8f acc0 = {}, acc1 = {};
  #pragma unroll 3
  for (int kk = 0; kk < Lq; kk += 32) {
    Frag16 a, b0, b1;
    a.q[0]  = *(const uint4*)(pa + kk);
    a.q[1]  = *(const uint4*)(pa + kk + 16);
    b0.q[0] = *(const uint4*)(pb0 + kk);
    b0.q[1] = *(const uint4*)(pb0 + kk + 16);
    b1.q[0] = *(const uint4*)(pb1 + kk);
    b1.q[1] = *(const uint4*)(pb1 + kk + 16);
    acc0 = wmma_f16(a.v, b0.v, acc0);
    acc1 = wmma_f16(a.v, b1.v, acc1);
  }

  #pragma unroll
  for (int r = 0; r < 8; ++r) {
    int i = it * 16 + r + ((lane >> 4) << 3);
    long base = ((long)n * Lq + i) * DP + h * DH;
    long idx0 = base + dl;
    long idx1 = base + 16 + dl;
    go[idx0] = (_Float16)(acc0[r] * gate[idx0]);
    go[idx1] = (_Float16)(acc1[r] * gate[idx1]);
  }
}

// ---------------------------------------------------------------------------
// 7) out = go @ Wo + bo (Wo transposed in LDS), transpose (n,i)->(i,n) store.
//    32-row tiles; n index is block-uniform (no per-lane division).
// ---------------------------------------------------------------------------
__global__ void k_out(const _Float16* __restrict__ go,
                      const float* __restrict__ Wo,
                      const float* __restrict__ bo,
                      float* __restrict__ out) {
  __shared__ _Float16 Wl[DP * DP];           // transposed: Wl[n*128 + k]
  for (int t = threadIdx.x; t < DP * DP; t += blockDim.x) {
    int k = t >> 7, n = t & 127;
    Wl[n * DP + k] = (_Float16)Wo[t];
  }
  __syncthreads();

  int lane = threadIdx.x & 31;
  long row0 = (long)blockIdx.x * 32;
  int n0 = (threadIdx.x >> 5) * 16;
  const _Float16* wb = Wl + (long)(n0 + (lane & 15)) * DP + ((lane >> 4) << 3);

  v8f acc0 = {}, acc1 = {};
  #pragma unroll
  for (int kk = 0; kk < DP; kk += 32) {
    v16h a0 = loadA(go + row0 * DP + kk, DP);
    v16h a1 = loadA(go + (row0 + 16) * DP + kk, DP);
    Frag16 b;
    b.q[0] = *(const uint4*)(wb + kk);
    b.q[1] = *(const uint4*)(wb + kk + 16);
    acc0 = wmma_f16(a0, b.v, acc0);
    acc1 = wmma_f16(a1, b.v, acc1);
  }

  const int nidx  = (int)(row0 / Lq);         // block-uniform
  const int ibase = (int)(row0 % Lq);
  const int col = n0 + (lane & 15);
  const float bias_c = bo[col];

  #pragma unroll
  for (int half = 0; half < 2; ++half) {
    v8f acc = half ? acc1 : acc0;
    #pragma unroll
    for (int r = 0; r < 8; ++r) {
      int iidx = ibase + half * 16 + r + ((lane >> 4) << 3);
      out[((long)iidx * Lq + nidx) * DP + col] = acc[r] + bias_c;
    }
  }
}

// ---------------------------------------------------------------------------
extern "C" void kernel_launch(void* const* d_in, const int* in_sizes, int n_in,
                              void* d_out, int out_size, void* d_ws, size_t ws_size,
                              hipStream_t stream) {
  const float* pair      = (const float*)d_in[0];
  const float* bias      = (const float*)d_in[1];
  const float* ln_pair_g = (const float*)d_in[2];
  const float* ln_pair_b = (const float*)d_in[3];
  const float* ln_bias_g = (const float*)d_in[4];
  const float* ln_bias_b = (const float*)d_in[5];
  const float* Wq        = (const float*)d_in[6];
  const float* Wk        = (const float*)d_in[7];
  const float* Wv        = (const float*)d_in[8];
  const float* Wb        = (const float*)d_in[9];
  const float* Wg        = (const float*)d_in[10];
  const float* bg        = (const float*)d_in[11];
  const float* Wo        = (const float*)d_in[12];
  const float* bo        = (const float*)d_in[13];
  float* out = (float*)d_out;

  // ---- workspace layout (all 16B-aligned) ----
  const size_t SZ_H = (size_t)NROW * DP * sizeof(_Float16);  // 37,748,736 B
  char* ws = (char*)d_ws;
  _Float16* P    = (_Float16*)(ws);
  _Float16* Qs   = (_Float16*)(ws + SZ_H);
  _Float16* Ks   = (_Float16*)(ws + 2 * SZ_H);
  _Float16* Vt   = (_Float16*)(ws + 3 * SZ_H);
  _Float16* go   = (_Float16*)(ws + 4 * SZ_H);
  float*    gate = (float*)   (ws + 5 * SZ_H);
  char* p2 = ws + 5 * SZ_H + (size_t)NROW * DP * sizeof(float);
  float*    bprj   = (float*)p2;                                   // L*L*4 f32
  float*    logits = (float*)(p2 + (size_t)Lq * Lq * NH * sizeof(float));
  _Float16* attn   = (_Float16*)(p2 + 2 * (size_t)Lq * Lq * NH * sizeof(float));

  const dim3 blk(256);

  // 1-2) LayerNorms (wave-per-row)
  k_ln_pair<<<dim3(NROW / 8), blk, 0, stream>>>(pair, ln_pair_g, ln_pair_b, P);
  k_ln_bias<<<dim3(NROW / 8), blk, 0, stream>>>(bias, ln_bias_g, ln_bias_b, Wb, bprj);

  // 3) Q/K/V/gate projections (4608 blocks of 32 rows x 8 waves over N=128)
  k_proj<<<dim3(NROW / 32), blk, 0, stream>>>(P, Wq, bg, (void*)Qs, 0);
  k_proj<<<dim3(NROW / 32), blk, 0, stream>>>(P, Wk, bg, (void*)Ks, 1);
  k_proj<<<dim3(NROW / 32), blk, 0, stream>>>(P, Wv, bg, (void*)Vt, 2);
  k_proj<<<dim3(NROW / 32), blk, 0, stream>>>(P, Wg, bg, (void*)gate, 3);

  // 4) logits: 4 heads x 12x24 (32x16) tiles = 1152 waves
  k_logits<<<dim3(1152 / 8), blk, 0, stream>>>(Qs, Ks, bprj, logits);

  // 5) softmax over j: NH*L = 1536 rows
  k_softmax<<<dim3(1536 / 8), blk, 0, stream>>>(logits, attn);

  // 6) attn @ V with fused gate: 4*384*24 = 36864 waves (16x32 tiles)
  k_av<<<dim3(36864 / 8), blk, 0, stream>>>(attn, Vt, gate, go);

  // 7) output projection + bias + transpose store
  k_out<<<dim3(NROW / 32), blk, 0, stream>>>(go, Wo, bo, out);
}
